// PaliGemmaWithExpertModel_49761491091928
// MI455X (gfx1250) — matrix operand order repaired and checked by reading, not compiled, not run on hardware
//
#include <hip/hip_runtime.h>
#include <math.h>

// ---------------------------------------------------------------------------
// Model constants (from reference)
// ---------------------------------------------------------------------------
static constexpr int kL  = 4;
static constexpr int kH  = 8;
static constexpr int kHD = 128;
static constexpr int kDV = 1024, kMV = 4096;
static constexpr int kDE = 512,  kME = 2048;
static constexpr int kB  = 2;
static constexpr int kS1 = 1024, kS2 = 64, kS = kS1 + kS2;
static constexpr float kEPS   = 1e-6f;
static constexpr float kSCALE = 0.08838834764831845f; // 128^-0.5

typedef __bf16        v16bf __attribute__((ext_vector_type(16)));
typedef float         v8f   __attribute__((ext_vector_type(8)));
typedef unsigned int  u32x4 __attribute__((ext_vector_type(4)));

__device__ __forceinline__ unsigned short f2bf(float f) {
    unsigned int u = __builtin_bit_cast(unsigned int, f);
    unsigned int r = u + 0x7FFFu + ((u >> 16) & 1u);   // round-to-nearest-even
    return (unsigned short)(r >> 16);
}

// ---------------------------------------------------------------------------
// Tiled bf16 WMMA GEMM: C[M,N] (f32) = alpha * A[M,K](bf16) * B[K,N](bf16)
// TRANSB=1 -> B given as Bt[N,K] row-major (used for Q*K^T).
// EPI: 0 = store, 1 = accumulate (+=) for residual adds.
// Per-z operand offset: (z / zdiv) * bstride + (z % zdiv) * hstride.
// Assumes: K % 64 == 0, N % 64 == 0, all operand leading dims (in bytes)
// multiples of 16B (true for this model) so staging is pure b128 traffic.
//
// Pipeline: double-buffered LDS, register-staged. Per K-step:
//   load tile t+1 (global b128) -> compute tile t (ds_load_b128 + WMMA)
//   -> store tile t+1 to other LDS buffer -> one barrier.
// LDS layouts are fragment-native so every WMMA operand is two ds_load_b128:
//   sA[m][k] row-major, sB[n][k] column-major, stride 72 ushorts (144B).
// ---------------------------------------------------------------------------
template <bool TRANSB, int EPI>
__global__ __launch_bounds__(256) void gemm_wmma(
    const unsigned short* __restrict__ A, int lda, long long a_bs, long long a_hs, int a_zd,
    const unsigned short* __restrict__ Bm, int ldb, long long b_bs, long long b_hs, int b_zd,
    float* __restrict__ C, int ldc, long long c_bs, long long c_hs, int c_zd,
    int M, int N, int K, float alpha)
{
    constexpr int BM = 128, BN = 64, BK = 64;
    constexpr int LSA = BK + 8;   // 72 ushorts = 144B: multiple of 16B
    constexpr int LSB = BK + 8;
    __shared__ unsigned short sA[2][BM][LSA];
    __shared__ unsigned short sB[2][BN][LSB];

    const int tid  = threadIdx.x;
    const int z    = blockIdx.z;
    const int wave = tid >> 5;
    const int lane = tid & 31;

    const unsigned short* Ab = A  + (size_t)((long long)(z / a_zd) * a_bs + (long long)(z % a_zd) * a_hs);
    const unsigned short* Bb = Bm + (size_t)((long long)(z / b_zd) * b_bs + (long long)(z % b_zd) * b_hs);
    float*                Cb = C  + (size_t)((long long)(z / c_zd) * c_bs + (long long)(z % c_zd) * c_hs);

    const int m0 = blockIdx.y * BM;
    const int n0 = blockIdx.x * BN;
    const bool interior = (m0 + BM <= M);   // block-uniform guard elision

    union { v8f v; float f[8]; } acc[4];
#pragma unroll
    for (int j = 0; j < 4; ++j)
#pragma unroll
        for (int r = 0; r < 8; ++r) acc[j].f[r] = 0.f;

    // fragment coordinates (constant over K loop)
    const int am  = (wave << 4) + (lane & 15);  // A row within tile
    const int akb = (lane >> 4) << 3;           // A k sub-offset {0,8}
    const int bn  = lane & 15;                  // B column within 16-group
    const int bkb = (lane >> 4) << 3;           // B k sub-offset {0,8}

    u32x4 ra[4];     // A tile stage: 1024 b128 / 256 thr = 4 each
    u32x4 rb[2];     // B tile stage

    auto loadA = [&](int k0) {
        if (interior) {
#pragma unroll
            for (int it = 0; it < 4; ++it) {
                int li = tid + it * 256;
                int r = li >> 3, q = li & 7;    // 8 b128 per 64-elem row
                ra[it] = *(const u32x4*)(Ab + (size_t)(m0 + r) * lda + k0 + q * 8);
            }
        } else {
#pragma unroll
            for (int it = 0; it < 4; ++it) {
                int li = tid + it * 256;
                int r = li >> 3, q = li & 7;
                int grow = m0 + r;
                u32x4 v = {0u, 0u, 0u, 0u};
                if (grow < M) v = *(const u32x4*)(Ab + (size_t)grow * lda + k0 + q * 8);
                ra[it] = v;
            }
        }
    };
    auto loadB = [&](int k0) {
        if (TRANSB) {
            // Bt[N,K]: rows are k-contiguous, direct b128
#pragma unroll
            for (int it = 0; it < 2; ++it) {
                int li = tid + it * 256;
                int n = li >> 3, q = li & 7;
                rb[it] = *(const u32x4*)(Bb + (size_t)(n0 + n) * ldb + k0 + q * 8);
            }
        } else {
            // B[K,N] row-major: thread owns a (k-pair, n-octet) 2x8 patch
            int kk2 = tid >> 3, oct = tid & 7;
            const unsigned short* p0 = Bb + (size_t)(k0 + kk2 * 2) * ldb + n0 + oct * 8;
            rb[0] = *(const u32x4*)(p0);
            rb[1] = *(const u32x4*)(p0 + ldb);
        }
    };
    auto storeAB = [&](int buf) {
#pragma unroll
        for (int it = 0; it < 4; ++it) {
            int li = tid + it * 256;
            int r = li >> 3, q = li & 7;
            *(u32x4*)&sA[buf][r][q * 8] = ra[it];
        }
        if (TRANSB) {
#pragma unroll
            for (int it = 0; it < 2; ++it) {
                int li = tid + it * 256;
                int n = li >> 3, q = li & 7;
                *(u32x4*)&sB[buf][n][q * 8] = rb[it];
            }
        } else {
            int kk2 = tid >> 3, oct = tid & 7;
            const unsigned int* r0 = (const unsigned int*)&rb[0];
            const unsigned int* r1 = (const unsigned int*)&rb[1];
#pragma unroll
            for (int p = 0; p < 4; ++p) {
                unsigned int d0 = r0[p], d1 = r1[p];
                unsigned int c0 = (d0 & 0xFFFFu) | (d1 << 16);          // column n
                unsigned int c1 = (d0 >> 16)     | (d1 & 0xFFFF0000u);  // column n+1
                int nl = oct * 8 + p * 2;
                *(unsigned int*)&sB[buf][nl + 0][kk2 * 2] = c0;
                *(unsigned int*)&sB[buf][nl + 1][kk2 * 2] = c1;
            }
        }
    };
    auto compute = [&](int buf) {
#pragma unroll
        for (int c = 0; c < 2; ++c) {
            union { v16bf v; u32x4 q[2]; } af;
            af.q[0] = *(const u32x4*)&sA[buf][am][c * 32 + akb];
            af.q[1] = *(const u32x4*)&sA[buf][am][c * 32 + 16 + akb];
#pragma unroll
            for (int j = 0; j < 4; ++j) {
                union { v16bf v; u32x4 q[2]; } bfg;
                const unsigned short* bp = &sB[buf][(j << 4) + bn][c * 32 + bkb];
                bfg.q[0] = *(const u32x4*)(bp);
                bfg.q[1] = *(const u32x4*)(bp + 16);
                acc[j].v = __builtin_amdgcn_wmma_f32_16x16x32_bf16(
                    false, af.v, false, bfg.v, (short)0, acc[j].v, false, false);
            }
        }
    };

    const int nk = K / BK;      // >= 2 for all GEMMs in this model
    loadA(0); loadB(0); storeAB(0);
    __syncthreads();
    int buf = 0;
    for (int kt = 0; kt < nk; ++kt) {
        const bool more = (kt + 1 < nk);
        if (more) { loadA((kt + 1) * BK); loadB((kt + 1) * BK); }
        compute(buf);
        if (more) storeAB(buf ^ 1);
        __syncthreads();
        buf ^= 1;
    }

    // ---- epilogue: VGPR r -> M=r (lanes 0-15) / M=8+r (lanes 16-31), N = lane&15
    const int crow = m0 + (wave << 4) + ((lane >> 4) << 3);
    const int ccol = n0 + (lane & 15);
    if (interior) {
#pragma unroll
        for (int j = 0; j < 4; ++j)
#pragma unroll
            for (int r = 0; r < 8; ++r) {
                size_t o = (size_t)(crow + r) * ldc + ccol + (j << 4);
                if (EPI == 0) Cb[o] = acc[j].f[r] * alpha;
                else          Cb[o] += acc[j].f[r] * alpha;
            }
    } else {
#pragma unroll
        for (int j = 0; j < 4; ++j)
#pragma unroll
            for (int r = 0; r < 8; ++r) {
                int row = crow + r;
                if (row < M) {
                    size_t o = (size_t)row * ldc + ccol + (j << 4);
                    if (EPI == 0) Cb[o] = acc[j].f[r] * alpha;
                    else          Cb[o] += acc[j].f[r] * alpha;
                }
            }
    }
}

// ---------------------------------------------------------------------------
// Elementwise / reduction kernels (f32 math, bf16 outputs feeding WMMA)
// ---------------------------------------------------------------------------
__global__ __launch_bounds__(256) void cvt_bf16_kernel(
    const float* __restrict__ x, unsigned short* __restrict__ y, long long n)
{
    long long i = (long long)blockIdx.x * 256 + threadIdx.x;
    if (i < n) y[i] = f2bf(x[i]);
}

__global__ __launch_bounds__(256) void rmsnorm_kernel(
    const float* __restrict__ x, const float* __restrict__ w,
    unsigned short* __restrict__ out, int D)
{
    const int row = blockIdx.x;
    const float* xr = x + (size_t)row * D;
    unsigned short* orow = out + (size_t)row * D;
    const int tid = threadIdx.x;

    float s = 0.f;
    for (int i = tid; i < D; i += 256) { float v = xr[i]; s += v * v; }
#pragma unroll
    for (int o = 16; o > 0; o >>= 1) s += __shfl_down(s, o, 32);

    __shared__ float red[8];
    if ((tid & 31) == 0) red[tid >> 5] = s;
    __syncthreads();
    if (tid == 0) {
        float t = 0.f;
        for (int j = 0; j < 8; ++j) t += red[j];
        red[0] = rsqrtf(t / (float)D + kEPS);
    }
    __syncthreads();
    const float rs = red[0];
    for (int i = tid; i < D; i += 256) orow[i] = f2bf(xr[i] * rs * (1.f + w[i]));
}

// RoPE over [rows, NH*128] f32 -> bf16; grid = (rows, NH), block = 64
__global__ void rope_kernel(const float* __restrict__ x, const int* __restrict__ pos,
                            unsigned short* __restrict__ out, int NH)
{
    const int row = blockIdx.x;
    const int h   = blockIdx.y;
    const int d   = threadIdx.x;                 // 0..63
    const float p = (float)pos[row];
    const float invf = expf(-(logf(10000.0f) * (float)(2 * d) / 128.0f));
    const float ang = p * invf;
    const float c = cosf(ang), sn = sinf(ang);
    const size_t base = ((size_t)row * NH + h) * 128;
    const float x0 = x[base + d], x1 = x[base + 64 + d];
    out[base + d]      = f2bf(x0 * c - x1 * sn);
    out[base + 64 + d] = f2bf(x1 * c + x0 * sn);
}

// f32 softmax with additive mask -> bf16 probs; one block per score row.
__global__ __launch_bounds__(256) void softmax_kernel(
    const float* __restrict__ sc, const float* __restrict__ mask,
    unsigned short* __restrict__ pr, int Slen, int HS)
{
    const int row = blockIdx.x;                  // z*S + q, z = b*H + h
    const int b   = row / HS;
    const int q   = row % Slen;
    const float* srow = sc + (size_t)row * Slen;
    const float* mrow = mask + ((size_t)b * Slen + q) * Slen;
    unsigned short* prow = pr + (size_t)row * Slen;
    const int tid = threadIdx.x;

    float v[8];
    float mx = -3.4e38f;
#pragma unroll
    for (int it = 0; it < 8; ++it) {
        int i = tid + it * 256;
        if (i < Slen) { v[it] = srow[i] + mrow[i]; mx = fmaxf(mx, v[it]); }
        else v[it] = -3.4e38f;
    }
#pragma unroll
    for (int o = 16; o > 0; o >>= 1) mx = fmaxf(mx, __shfl_xor(mx, o, 32));

    __shared__ float red[8];
    __shared__ float bcast;
    if ((tid & 31) == 0) red[tid >> 5] = mx;
    __syncthreads();
    if (tid == 0) {
        float t = red[0];
        for (int j = 1; j < 8; ++j) t = fmaxf(t, red[j]);
        bcast = t;
    }
    __syncthreads();
    mx = bcast;

    float s = 0.f;
#pragma unroll
    for (int it = 0; it < 8; ++it) {
        int i = tid + it * 256;
        if (i < Slen) { v[it] = __expf(v[it] - mx); s += v[it]; }
    }
#pragma unroll
    for (int o = 16; o > 0; o >>= 1) s += __shfl_xor(s, o, 32);
    if ((tid & 31) == 0) red[tid >> 5] = s;
    __syncthreads();
    if (tid == 0) {
        float t = 0.f;
        for (int j = 0; j < 8; ++j) t += red[j];
        bcast = t;
    }
    __syncthreads();
    const float inv = 1.f / bcast;
#pragma unroll
    for (int it = 0; it < 8; ++it) {
        int i = tid + it * 256;
        if (i < Slen) prow[i] = f2bf(v[it] * inv);
    }
}

// X = gelu_tanh(G) * U  -> bf16
__global__ __launch_bounds__(256) void gelugate_kernel(
    const float* __restrict__ G, const float* __restrict__ U,
    unsigned short* __restrict__ X, long long n)
{
    long long i = (long long)blockIdx.x * 256 + threadIdx.x;
    if (i < n) {
        float g  = G[i];
        float t  = tanhf(0.7978845608028654f * (g + 0.044715f * g * g * g));
        X[i] = f2bf(0.5f * g * (1.f + t) * U[i]);
    }
}

// ---------------------------------------------------------------------------
// Host-side dispatch
// ---------------------------------------------------------------------------
static inline void run_gemm(hipStream_t st, bool transb, int epi,
    const unsigned short* A, int lda, long long a_bs, long long a_hs, int a_zd,
    const unsigned short* Bm, int ldb, long long b_bs, long long b_hs, int b_zd,
    float* C, int ldc, long long c_bs, long long c_hs, int c_zd,
    int M, int N, int K, int Z, float alpha)
{
    dim3 grid((unsigned)(N / 64), (unsigned)((M + 127) / 128), (unsigned)Z);
    dim3 block(256);
    if (!transb) {
        if (epi == 0)
            gemm_wmma<false, 0><<<grid, block, 0, st>>>(A, lda, a_bs, a_hs, a_zd,
                Bm, ldb, b_bs, b_hs, b_zd, C, ldc, c_bs, c_hs, c_zd, M, N, K, alpha);
        else
            gemm_wmma<false, 1><<<grid, block, 0, st>>>(A, lda, a_bs, a_hs, a_zd,
                Bm, ldb, b_bs, b_hs, b_zd, C, ldc, c_bs, c_hs, c_zd, M, N, K, alpha);
    } else {
        gemm_wmma<true, 0><<<grid, block, 0, st>>>(A, lda, a_bs, a_hs, a_zd,
            Bm, ldb, b_bs, b_hs, b_zd, C, ldc, c_bs, c_hs, c_zd, M, N, K, alpha);
    }
}

static inline void run_cvt(hipStream_t st, const float* x, unsigned short* y, long long n) {
    cvt_bf16_kernel<<<(unsigned)((n + 255) / 256), 256, 0, st>>>(x, y, n);
}

extern "C" void kernel_launch(void* const* d_in, const int* in_sizes, int n_in,
                              void* d_out, int out_size, void* d_ws, size_t ws_size,
                              hipStream_t stream)
{
    (void)in_sizes; (void)n_in; (void)out_size; (void)ws_size;

    const float* emb_v = (const float*)d_in[0];
    const float* emb_e = (const float*)d_in[1];
    const float* mask  = (const float*)d_in[2];
    const int*   pos   = (const int*)d_in[3];
    const float* pv[9]; for (int i = 0; i < 9; ++i) pv[i] = (const float*)d_in[4 + i];
    const float* pe[9]; for (int i = 0; i < 9; ++i) pe[i] = (const float*)d_in[13 + i];
    // param order per stream: 0=ln1 1=wq 2=wk 3=wv 4=wo 5=ln2 6=wg 7=wu 8=wd

    float* out_v = (float*)d_out;
    float* out_e = out_v + (size_t)kB * kS1 * kDV;

    // ---- workspace bump allocator ----
    size_t cursor = 0;
    char*  wsb    = (char*)d_ws;
    auto alloc = [&](size_t bytes) -> char* {
        char* p = wsb + cursor;
        cursor += (bytes + 255) & ~(size_t)255;
        return p;
    };

    // ---- bf16 weight copies (per full L-stack) ----
    static const int    widx[7]  = {1, 2, 3, 4, 6, 7, 8};
    static const size_t wszv[7]  = {(size_t)kDV*kH*kHD, (size_t)kDV*kHD, (size_t)kDV*kHD,
                                    (size_t)kH*kHD*kDV, (size_t)kDV*kMV, (size_t)kDV*kMV,
                                    (size_t)kMV*kDV};
    static const size_t wsze[7]  = {(size_t)kDE*kH*kHD, (size_t)kDE*kHD, (size_t)kDE*kHD,
                                    (size_t)kH*kHD*kDE, (size_t)kDE*kME, (size_t)kDE*kME,
                                    (size_t)kME*kDE};
    unsigned short* wbv[7];
    unsigned short* wbe[7];
    for (int i = 0; i < 7; ++i) {
        wbv[i] = (unsigned short*)alloc(kL * wszv[i] * 2);
        run_cvt(stream, pv[widx[i]], wbv[i], (long long)kL * wszv[i]);
    }
    for (int i = 0; i < 7; ++i) {
        wbe[i] = (unsigned short*)alloc(kL * wsze[i] * 2);
        run_cvt(stream, pe[widx[i]], wbe[i], (long long)kL * wsze[i]);
    }

    // ---- activation scratch ----
    unsigned short* h_v  = (unsigned short*)alloc((size_t)kB * kS1 * kDV * 2);
    unsigned short* h_e  = (unsigned short*)alloc((size_t)kB * kS2 * kDE * 2);
    float* qf  = (float*)alloc((size_t)kB * kS * kH * kHD * 4);
    float* kf  = (float*)alloc((size_t)kB * kS * kHD * 4);
    float* vf  = (float*)alloc((size_t)kB * kS * kHD * 4);
    unsigned short* qb = (unsigned short*)alloc((size_t)kB * kS * kH * kHD * 2);
    unsigned short* kb = (unsigned short*)alloc((size_t)kB * kS * kHD * 2);
    unsigned short* vb = (unsigned short*)alloc((size_t)kB * kS * kHD * 2);
    float* sc  = (float*)alloc((size_t)kB * kH * kS * kS * 4);
    unsigned short* pb = (unsigned short*)alloc((size_t)kB * kH * kS * kS * 2);
    float* att = (float*)alloc((size_t)kB * kS * kH * kHD * 4);
    unsigned short* attb = (unsigned short*)alloc((size_t)kB * kS * kH * kHD * 2);
    float* Gf  = (float*)alloc((size_t)kB * kS1 * kMV * 4);
    float* Uf  = (float*)alloc((size_t)kB * kS1 * kMV * 4);
    unsigned short* Xb = (unsigned short*)alloc((size_t)kB * kS1 * kMV * 2);

    // ---- residual streams live in d_out ----
    hipMemcpyAsync(out_v, emb_v, (size_t)kB * kS1 * kDV * 4, hipMemcpyDeviceToDevice, stream);
    hipMemcpyAsync(out_e, emb_e, (size_t)kB * kS2 * kDE * 4, hipMemcpyDeviceToDevice, stream);

    const long long qRow = (long long)kS * kH * kHD;   // 1088*1024
    const long long kRow = (long long)kS * kHD;        // 1088*128

    for (int li = 0; li < kL; ++li) {
        const unsigned short* wq_v = wbv[0] + (size_t)li * wszv[0];
        const unsigned short* wk_v = wbv[1] + (size_t)li * wszv[1];
        const unsigned short* wv_v = wbv[2] + (size_t)li * wszv[2];
        const unsigned short* wo_v = wbv[3] + (size_t)li * wszv[3];
        const unsigned short* wg_v = wbv[4] + (size_t)li * wszv[4];
        const unsigned short* wu_v = wbv[5] + (size_t)li * wszv[5];
        const unsigned short* wd_v = wbv[6] + (size_t)li * wszv[6];
        const unsigned short* wq_e = wbe[0] + (size_t)li * wsze[0];
        const unsigned short* wk_e = wbe[1] + (size_t)li * wsze[1];
        const unsigned short* wv_e = wbe[2] + (size_t)li * wsze[2];
        const unsigned short* wo_e = wbe[3] + (size_t)li * wsze[3];
        const unsigned short* wg_e = wbe[4] + (size_t)li * wsze[4];
        const unsigned short* wu_e = wbe[5] + (size_t)li * wsze[5];
        const unsigned short* wd_e = wbe[6] + (size_t)li * wsze[6];

        // ---- input RMSNorm per stream ----
        rmsnorm_kernel<<<kB * kS1, 256, 0, stream>>>(out_v, pv[0] + (size_t)li * kDV, h_v, kDV);
        rmsnorm_kernel<<<kB * kS2, 256, 0, stream>>>(out_e, pe[0] + (size_t)li * kDE, h_e, kDE);

        // ---- QKV projections; write into concatenated [B, S, *] buffers ----
        // VLM rows -> rows [0,S1) of each batch; expert rows -> [S1,S)
        run_gemm(stream, false, 0, h_v, kDV, (long long)kS1 * kDV, 0, 1,
                 wq_v, kH * kHD, 0, 0, 1, qf, kH * kHD, qRow, 0, 1,
                 kS1, kH * kHD, kDV, kB, 1.f);
        run_gemm(stream, false, 0, h_v, kDV, (long long)kS1 * kDV, 0, 1,
                 wk_v, kHD, 0, 0, 1, kf, kHD, kRow, 0, 1, kS1, kHD, kDV, kB, 1.f);
        run_gemm(stream, false, 0, h_v, kDV, (long long)kS1 * kDV, 0, 1,
                 wv_v, kHD, 0, 0, 1, vf, kHD, kRow, 0, 1, kS1, kHD, kDV, kB, 1.f);
        run_gemm(stream, false, 0, h_e, kDE, (long long)kS2 * kDE, 0, 1,
                 wq_e, kH * kHD, 0, 0, 1, qf + (size_t)kS1 * kH * kHD, kH * kHD, qRow, 0, 1,
                 kS2, kH * kHD, kDE, kB, 1.f);
        run_gemm(stream, false, 0, h_e, kDE, (long long)kS2 * kDE, 0, 1,
                 wk_e, kHD, 0, 0, 1, kf + (size_t)kS1 * kHD, kHD, kRow, 0, 1,
                 kS2, kHD, kDE, kB, 1.f);
        run_gemm(stream, false, 0, h_e, kDE, (long long)kS2 * kDE, 0, 1,
                 wv_e, kHD, 0, 0, 1, vf + (size_t)kS1 * kHD, kHD, kRow, 0, 1,
                 kS2, kHD, kDE, kB, 1.f);

        // ---- RoPE (q: 8 heads, k: 1 kv head) + v cast ----
        rope_kernel<<<dim3(kB * kS, kH), 64, 0, stream>>>(qf, pos, qb, kH);
        rope_kernel<<<dim3(kB * kS, 1), 64, 0, stream>>>(kf, pos, kb, 1);
        run_cvt(stream, vf, vb, (long long)kB * kS * kHD);

        // ---- scores = Q * K^T * SCALE  (GQA: k shared across 8 heads via zdiv) ----
        run_gemm(stream, true, 0,
                 qb, kH * kHD, qRow, kHD, kH,               // A: per (b,h) strided view
                 kb, kHD, kRow, 0, kH,                      // Bt = K[b] (128-wide rows)
                 sc, kS, (long long)kS * kS, 0, 1,
                 kS, kS, kHD, kB * kH, kSCALE);

        // ---- masked f32 softmax -> bf16 probs ----
        softmax_kernel<<<kB * kH * kS, 256, 0, stream>>>(sc, mask, pb, kS, kH * kS);

        // ---- att = P * V, scattered back to [B, S, H*HD] ----
        run_gemm(stream, false, 0,
                 pb, kS, (long long)kS * kS, 0, 1,
                 vb, kHD, kRow, 0, kH,
                 att, kH * kHD, qRow, kHD, kH,
                 kS, kHD, kS, kB * kH, 1.f);
        run_cvt(stream, att, attb, (long long)kB * kS * kH * kHD);

        // ---- o_proj + residual add ----
        run_gemm(stream, false, 1,
                 attb, kH * kHD, qRow, 0, 1,
                 wo_v, kDV, 0, 0, 1,
                 out_v, kDV, (long long)kS1 * kDV, 0, 1,
                 kS1, kDV, kH * kHD, kB, 1.f);
        run_gemm(stream, false, 1,
                 attb + (size_t)kS1 * kH * kHD, kH * kHD, qRow, 0, 1,
                 wo_e, kDE, 0, 0, 1,
                 out_e, kDE, (long long)kS2 * kDE, 0, 1,
                 kS2, kDE, kH * kHD, kB, 1.f);

        // ---- MLP, VLM stream (batch flattened: contiguous in/out) ----
        rmsnorm_kernel<<<kB * kS1, 256, 0, stream>>>(out_v, pv[5] + (size_t)li * kDV, h_v, kDV);
        run_gemm(stream, false, 0, h_v, kDV, 0, 0, 1, wg_v, kMV, 0, 0, 1,
                 Gf, kMV, 0, 0, 1, kB * kS1, kMV, kDV, 1, 1.f);
        run_gemm(stream, false, 0, h_v, kDV, 0, 0, 1, wu_v, kMV, 0, 0, 1,
                 Uf, kMV, 0, 0, 1, kB * kS1, kMV, kDV, 1, 1.f);
        {
            long long n = (long long)kB * kS1 * kMV;
            gelugate_kernel<<<(unsigned)((n + 255) / 256), 256, 0, stream>>>(Gf, Uf, Xb, n);
        }
        run_gemm(stream, false, 1, Xb, kMV, 0, 0, 1, wd_v, kDV, 0, 0, 1,
                 out_v, kDV, 0, 0, 1, kB * kS1, kDV, kMV, 1, 1.f);

        // ---- MLP, expert stream ----
        rmsnorm_kernel<<<kB * kS2, 256, 0, stream>>>(out_e, pe[5] + (size_t)li * kDE, h_e, kDE);
        run_gemm(stream, false, 0, h_e, kDE, 0, 0, 1, wg_e, kME, 0, 0, 1,
                 Gf, kME, 0, 0, 1, kB * kS2, kME, kDE, 1, 1.f);
        run_gemm(stream, false, 0, h_e, kDE, 0, 0, 1, wu_e, kME, 0, 0, 1,
                 Uf, kME, 0, 0, 1, kB * kS2, kME, kDE, 1, 1.f);
        {
            long long n = (long long)kB * kS2 * kME;
            gelugate_kernel<<<(unsigned)((n + 255) / 256), 256, 0, stream>>>(Gf, Uf, Xb, n);
        }
        run_gemm(stream, false, 1, Xb, kME, 0, 0, 1, wd_e, kDE, 0, 0, 1,
                 out_e, kDE, 0, 0, 1, kB * kS2, kDE, kME, 1, 1.f);
    }
}
